// SS2D_52561809769032
// MI455X (gfx1250) — matrix operand order repaired and checked by reading, not compiled, not run on hardware
//
#include <hip/hip_runtime.h>
#include <hip/hip_bf16.h>
#include <math.h>

// ---------------------------------------------------------------------------
// SS2D (VMamba selective-scan 2D) forward for gfx1250 / MI455X.
//   GEMMs -> v_wmma_f32_16x16x32_f16, 16x64 wave tile (A-fragment reused 4x)
//   scan  -> f32 VALU with LDS-staged x_dbl chunks, atomic direction merge
// ---------------------------------------------------------------------------

typedef __attribute__((ext_vector_type(16))) _Float16 v16h;
typedef __attribute__((ext_vector_type(8)))  _Float16 v8h;
typedef __attribute__((ext_vector_type(8)))  float    v8f;

#define BB 4
#define HH 64
#define WW 64
#define DM 768
#define DI 384
#define NS 16
#define RR 48
#define KD 4
#define LL 4096          // H*W
#define ML (BB*LL)       // 16384 rows

// cross-scan index map: xs[b,k,c,l] = xc[b,c,mapL(k,l)]; merge uses the same map
__device__ __forceinline__ int mapL(int k, int l) {
    int j = (k >= 2) ? (LL - 1 - l) : l;
    if (k & 1) j = ((j & 63) << 6) | (j >> 6);   // 64x64 transpose (involution)
    return j;
}

// ---------------------------------------------------------------------------
// f32 -> f16 conversion
// ---------------------------------------------------------------------------
__global__ void cvt_f32_f16(const float* __restrict__ in, _Float16* __restrict__ out, int n) {
    int i = blockIdx.x * blockDim.x + threadIdx.x;
    if (i < n) out[i] = (_Float16)in[i];
}

// ---------------------------------------------------------------------------
// WMMA GEMM:  C[M,N] = A[M,K] * W[N,K]^T   (row-major, f16 in, f32 out)
// Wave tile = 16(M) x 64(N): 4 accumulators; per 32-wide k-step one shared A
// fragment feeds 4 WMMAs (A traffic /4, wmma density x4 vs naive).
// A fragment (16x32 f16): lane holds row m=lane&15, halves cover
//   K = hi*16..hi*16+15                     -> two contiguous b128 loads.
// B fragment (32x16 f16): lane holds col n=lane&15, halves cover
//   K = hi*16..hi*16+15                     -> two contiguous b128 loads.
// C/D: VGPR r at (M = r + 8*hi, N = lane&15).
// Requires: N % 64 == 0, K % 32 == 0, M % 16 == 0.
// ---------------------------------------------------------------------------
__global__ void __launch_bounds__(256) gemm_f16_wmma(
    const _Float16* __restrict__ A, long long aStride, int aMod, int lda,
    const _Float16* __restrict__ Wt, long long wStride, int ldw,
    float* __restrict__ C, long long cStride, int ldc,
    int M, int N, int K)
{
    const int lane = threadIdx.x & 31;
    const int wave = blockIdx.x * (blockDim.x >> 5) + (threadIdx.x >> 5);
    const int n4t  = N >> 6;                 // 64-column groups
    const int mt   = wave / n4t;
    const int nt4  = wave - mt * n4t;
    if (mt >= (M >> 4)) return;

    const _Float16* Ab = A  + (size_t)(blockIdx.y % aMod) * aStride;
    const _Float16* Wb = Wt + (size_t)blockIdx.y * wStride;
    float*          Cb = C  + (size_t)blockIdx.y * cStride;

    const int hi = lane >> 4;
    const int m  = (mt << 4) + (lane & 15);
    const int n0 = (nt4 << 6) + (lane & 15);
    const _Float16* arow = Ab + (size_t)m * lda + hi * 16;
    const _Float16* wrow0 = Wb + (size_t)(n0     ) * ldw + hi * 16;
    const _Float16* wrow1 = Wb + (size_t)(n0 + 16) * ldw + hi * 16;
    const _Float16* wrow2 = Wb + (size_t)(n0 + 32) * ldw + hi * 16;
    const _Float16* wrow3 = Wb + (size_t)(n0 + 48) * ldw + hi * 16;

    v8f acc0 = {}, acc1 = {}, acc2 = {}, acc3 = {};
    for (int k0 = 0; k0 < K; k0 += 32) {
        union { v16h v; v8h h[2]; } a, b0, b1, b2, b3;
        a.h[0]  = *(const v8h*)(arow  + k0);
        a.h[1]  = *(const v8h*)(arow  + k0 + 8);
        b0.h[0] = *(const v8h*)(wrow0 + k0);
        b0.h[1] = *(const v8h*)(wrow0 + k0 + 8);
        b1.h[0] = *(const v8h*)(wrow1 + k0);
        b1.h[1] = *(const v8h*)(wrow1 + k0 + 8);
        b2.h[0] = *(const v8h*)(wrow2 + k0);
        b2.h[1] = *(const v8h*)(wrow2 + k0 + 8);
        b3.h[0] = *(const v8h*)(wrow3 + k0);
        b3.h[1] = *(const v8h*)(wrow3 + k0 + 8);
        __builtin_prefetch(arow + k0 + 32, 0, 0);   // global_prefetch_b8
        acc0 = __builtin_amdgcn_wmma_f32_16x16x32_f16(
                   false, a.v, false, b0.v, (short)0, acc0, false, false);
        acc1 = __builtin_amdgcn_wmma_f32_16x16x32_f16(
                   false, a.v, false, b1.v, (short)0, acc1, false, false);
        acc2 = __builtin_amdgcn_wmma_f32_16x16x32_f16(
                   false, a.v, false, b2.v, (short)0, acc2, false, false);
        acc3 = __builtin_amdgcn_wmma_f32_16x16x32_f16(
                   false, a.v, false, b3.v, (short)0, acc3, false, false);
    }

    float* crow = Cb + (size_t)((mt << 4) + hi * 8) * ldc + n0;
    #pragma unroll
    for (int r = 0; r < 8; ++r) {
        crow[(size_t)r * ldc     ] = acc0[r];
        crow[(size_t)r * ldc + 16] = acc1[r];
        crow[(size_t)r * ldc + 32] = acc2[r];
        crow[(size_t)r * ldc + 48] = acc3[r];
    }
}

// ---------------------------------------------------------------------------
// Depthwise 3x3 conv (SAME, zero pad) + bias + SiLU, then scatter the result
// into all four scan directions:  xs[b][k][l][c] (f16, row = l, contiguous c).
// One block per (b,l); thread = channel c  (coalesced reads of xz rows).
// ---------------------------------------------------------------------------
__global__ void __launch_bounds__(DI) conv_silu_scatter(
    const float* __restrict__ xz,      // [ML][2*DI]
    const float* __restrict__ conv_w,  // [DI][1][3][3]
    const float* __restrict__ conv_b,  // [DI]
    _Float16* __restrict__ xs)         // [B][K][LL][DI]
{
    const int c  = threadIdx.x;
    const int bl = blockIdx.x;         // b*LL + l
    const int b  = bl >> 12;
    const int l  = bl & (LL - 1);
    const int h  = l >> 6, w = l & 63;

    float acc = conv_b[c];
    #pragma unroll
    for (int dh = -1; dh <= 1; ++dh) {
        int hh = h + dh; if ((unsigned)hh >= (unsigned)HH) continue;
        #pragma unroll
        for (int dw = -1; dw <= 1; ++dw) {
            int ww = w + dw; if ((unsigned)ww >= (unsigned)WW) continue;
            float wv = conv_w[c * 9 + (dh + 1) * 3 + (dw + 1)];
            acc += wv * xz[((size_t)b * LL + hh * WW + ww) * (2 * DI) + c];
        }
    }
    float v = acc / (1.f + __expf(-acc));        // SiLU
    _Float16 hv = (_Float16)v;

    const int t = ((l & 63) << 6) | (l >> 6);    // transpose position
    const size_t base = (size_t)b * (KD * (size_t)LL * DI);
    xs[base + ((size_t)(0 * LL + l))           * DI + c] = hv;
    xs[base + ((size_t)(1 * LL + t))           * DI + c] = hv;
    xs[base + ((size_t)(2 * LL + (LL - 1 - l)))* DI + c] = hv;
    xs[base + ((size_t)(3 * LL + (LL - 1 - t)))* DI + c] = hv;
}

// ---------------------------------------------------------------------------
// Selective scan. One block per (b,k); thread = channel c. Sequential over L.
// Stages x_dbl[bk][0:80][l0:l0+32] in LDS (rows 0..47 = dt lowrank, 48..63 = B,
// 64..79 = C); dt is rebuilt per channel via a 48-FMA dot (avoids a 100MB dts
// buffer). Direction merge via global f32 atomics into ym[b][pos][c].
// ---------------------------------------------------------------------------
#define TL 32
__global__ void __launch_bounds__(DI) selective_scan_kernel(
    const float* __restrict__ xdbl,    // [16][80][LL]
    const _Float16* __restrict__ xs,   // [16][LL][DI]
    const float* __restrict__ dtw_g,   // [K][DI][RR]
    const float* __restrict__ dtb_g,   // [K][DI]
    const float* __restrict__ Alog,    // [K*DI][NS]
    const float* __restrict__ Dg,      // [K*DI]
    float* __restrict__ ym)            // [B][LL][DI]
{
    __shared__ float lds[80 * TL];
    const int c  = threadIdx.x;
    const int bk = blockIdx.x;         // b*4 + k
    const int b  = bk >> 2, k = bk & 3;
    const int kc = k * DI + c;

    float dtw[RR];
    #pragma unroll
    for (int r = 0; r < RR; ++r) dtw[r] = dtw_g[(size_t)kc * RR + r];
    float Av[NS];
    #pragma unroll
    for (int n = 0; n < NS; ++n) Av[n] = -__expf(Alog[(size_t)kc * NS + n]);
    const float Dv  = Dg[kc];
    const float dtb = dtb_g[kc];
    float h[NS];
    #pragma unroll
    for (int n = 0; n < NS; ++n) h[n] = 0.f;

    const float*    xb = xdbl + (size_t)bk * 80 * LL;
    const _Float16* ub = xs   + (size_t)bk * LL * DI;
    float*          yb = ym   + (size_t)b  * LL * DI;

    for (int l0 = 0; l0 < LL; l0 += TL) {
        __syncthreads();
        for (int i = threadIdx.x; i < 80 * TL; i += DI) {
            int row = i >> 5, li = i & 31;
            lds[i] = xb[(size_t)row * LL + l0 + li];
        }
        __syncthreads();
        for (int li = 0; li < TL; ++li) {
            const int l = l0 + li;
            float dtr = dtb;
            #pragma unroll
            for (int r = 0; r < RR; ++r) dtr += lds[r * TL + li] * dtw[r];
            float dt = (dtr > 20.f) ? dtr : log1pf(__expf(dtr));   // softplus
            float u  = (float)ub[(size_t)l * DI + c];
            float du = dt * u;
            float y = 0.f;
            #pragma unroll
            for (int n = 0; n < NS; ++n) {
                float Bv = lds[(RR + n) * TL + li];
                float Cv = lds[(RR + NS + n) * TL + li];
                h[n] = h[n] * __expf(dt * Av[n]) + du * Bv;
                y += h[n] * Cv;
            }
            y += Dv * u;
            atomicAdd(&yb[(size_t)mapL(k, l) * DI + c], y);
        }
    }
}

// ---------------------------------------------------------------------------
// LayerNorm over channels + SiLU gate, output f16 for the final WMMA GEMM.
// One block per (b,l); wave32 shuffle reduce + small LDS combine (12 waves).
// ---------------------------------------------------------------------------
__global__ void __launch_bounds__(DI) ln_gate_kernel(
    const float* __restrict__ ym,      // [ML][DI]
    const float* __restrict__ xz,      // [ML][2*DI] (z in upper half)
    const float* __restrict__ g, const float* __restrict__ bta,
    _Float16* __restrict__ yg)         // [ML][DI]
{
    __shared__ float ps[DI / 32], pss[DI / 32];
    const int c  = threadIdx.x;
    const int bl = blockIdx.x;
    float v  = ym[(size_t)bl * DI + c];
    float s = v, ss = v * v;
    #pragma unroll
    for (int off = 16; off > 0; off >>= 1) {
        s  += __shfl_down(s,  off, 32);
        ss += __shfl_down(ss, off, 32);
    }
    const int wid = c >> 5, lane = c & 31;
    if (lane == 0) { ps[wid] = s; pss[wid] = ss; }
    __syncthreads();
    if (c == 0) {
        float t = 0.f, tt = 0.f;
        #pragma unroll
        for (int i = 0; i < DI / 32; ++i) { t += ps[i]; tt += pss[i]; }
        float mu = t / (float)DI;
        ps[0]  = mu;
        pss[0] = tt / (float)DI - mu * mu;
    }
    __syncthreads();
    float mu  = ps[0];
    float inv = rsqrtf(pss[0] + 1e-5f);
    float yn  = (v - mu) * inv * g[c] + bta[c];
    float z   = xz[(size_t)bl * (2 * DI) + DI + c];
    float sz  = z / (1.f + __expf(-z));            // SiLU(z)
    yg[(size_t)bl * DI + c] = (_Float16)(yn * sz);
}

// ---------------------------------------------------------------------------
// Launcher
// ---------------------------------------------------------------------------
extern "C" void kernel_launch(void* const* d_in, const int* in_sizes, int n_in,
                              void* d_out, int out_size, void* d_ws, size_t ws_size,
                              hipStream_t stream)
{
    (void)in_sizes; (void)n_in; (void)out_size; (void)ws_size;

    const float* x          = (const float*)d_in[0];
    const float* in_proj_w  = (const float*)d_in[1];   // [2*DI][DM]
    const float* conv_w     = (const float*)d_in[2];
    const float* conv_b     = (const float*)d_in[3];
    const float* x_proj_w   = (const float*)d_in[4];   // [K][80][DI]
    const float* dt_projs_w = (const float*)d_in[5];   // [K][DI][RR]
    const float* dt_projs_b = (const float*)d_in[6];   // [K][DI]
    const float* A_logs     = (const float*)d_in[7];   // [K*DI][NS]
    const float* Ds         = (const float*)d_in[8];   // [K*DI]
    const float* out_norm_g = (const float*)d_in[9];
    const float* out_norm_b = (const float*)d_in[10];
    const float* out_proj_w = (const float*)d_in[11];  // [DM][DI]
    float* out = (float*)d_out;

    // workspace carve-up (~187 MB)
    char* ws = (char*)d_ws;
    size_t off = 0;
    auto alloc = [&](size_t bytes) {
        char* p = ws + off;
        off += (bytes + 255) & ~(size_t)255;
        return p;
    };
    _Float16* x_h    = (_Float16*)alloc((size_t)ML * DM * 2);
    _Float16* win_h  = (_Float16*)alloc((size_t)(2 * DI) * DM * 2);
    _Float16* wxp_h  = (_Float16*)alloc((size_t)KD * 80 * DI * 2);
    _Float16* wout_h = (_Float16*)alloc((size_t)DM * DI * 2);
    float*    xz     = (float*)   alloc((size_t)ML * (2 * DI) * 4);
    _Float16* xs     = (_Float16*)alloc((size_t)16 * LL * DI * 2);
    float*    xdbl   = (float*)   alloc((size_t)16 * 80 * LL * 4);
    float*    ym     = (float*)   alloc((size_t)ML * DI * 4);
    _Float16* yg     = (_Float16*)alloc((size_t)ML * DI * 2);

    // 1) f16 casts of activations + GEMM weights
    {
        int n;
        n = ML * DM;          cvt_f32_f16<<<(n + 255) / 256, 256, 0, stream>>>(x, x_h, n);
        n = 2 * DI * DM;      cvt_f32_f16<<<(n + 255) / 256, 256, 0, stream>>>(in_proj_w, win_h, n);
        n = KD * 80 * DI;     cvt_f32_f16<<<(n + 255) / 256, 256, 0, stream>>>(x_proj_w, wxp_h, n);
        n = DM * DI;          cvt_f32_f16<<<(n + 255) / 256, 256, 0, stream>>>(out_proj_w, wout_h, n);
    }

    // 2) in_proj: xz[ML,768] = x_h[ML,768] @ in_proj_w[768,768]^T
    {
        const int M = ML, N = 2 * DI, K = DM;
        const int waves = (M / 16) * (N / 64);           // 12288 -> 1536 blocks
        gemm_f16_wmma<<<dim3(waves / 8, 1), 256, 0, stream>>>(
            x_h, 0, 1, K, win_h, 0, K, xz, 0, N, M, N, K);
    }

    // 3) depthwise conv + SiLU + 4-direction scatter
    conv_silu_scatter<<<ML, DI, 0, stream>>>(xz, conv_w, conv_b, xs);

    // 4) x_dbl[bk][80][LL] = x_proj_w[k][80,384] @ xs[bk][LL,384]^T   (16 batches)
    {
        const int M = 80, N = LL, K = DI;
        const int waves = (M / 16) * (N / 64);           // 320 -> 40 blocks x 16
        gemm_f16_wmma<<<dim3(waves / 8, 16), 256, 0, stream>>>(
            wxp_h, (long long)80 * DI, KD, DI,           // A cycles over k = bk%4
            xs,    (long long)LL * DI, DI,
            xdbl,  (long long)80 * LL, LL, M, N, K);
    }

    // 5) selective scan + atomic direction merge
    (void)hipMemsetAsync(ym, 0, (size_t)ML * DI * 4, stream);
    selective_scan_kernel<<<16, DI, 0, stream>>>(
        xdbl, xs, dt_projs_w, dt_projs_b, A_logs, Ds, ym);

    // 6) LayerNorm + SiLU gate -> f16
    ln_gate_kernel<<<ML, DI, 0, stream>>>(ym, xz, out_norm_g, out_norm_b, yg);

    // 7) out_proj: out[ML,768] = yg[ML,384] @ out_proj_w[768,384]^T
    {
        const int M = ML, N = DM, K = DI;
        const int waves = (M / 16) * (N / 64);
        gemm_f16_wmma<<<dim3(waves / 8, 1), 256, 0, stream>>>(
            yg, 0, 1, K, wout_h, 0, K, out, 0, N, M, N, K);
    }
}
